// RefMistral4EagleMLA_31473520345298
// MI455X (gfx1250) — matrix operand order repaired and checked by reading, not compiled, not run on hardware
//
#include <hip/hip_runtime.h>

// ---------------------------------------------------------------------------
// MLA forward (DeepSeek-style) for MI455X / gfx1250.
// All matmuls via v_wmma_f32_16x16x32_bf16 (f32 accum). fp32 -> bf16 is done
// ONCE per tensor. GEMM tiles are staged with GLOBAL_LOAD_ASYNC_TO_LDS_B128
// (ASYNCcnt) and double-buffered so the copy of tile k+1 overlaps the WMMAs
// of tile k.
// ---------------------------------------------------------------------------

typedef __attribute__((ext_vector_type(8)))  float        v8f;
typedef __attribute__((ext_vector_type(16))) __bf16       v16bf;
typedef __attribute__((ext_vector_type(2)))  __bf16       v2bf;
typedef __attribute__((ext_vector_type(8)))  unsigned int u32x8;

__device__ __forceinline__ unsigned short bf16_bits(float x) {
  return __builtin_bit_cast(unsigned short, (__bf16)x);   // native cvt
}
__device__ __forceinline__ unsigned int pack_bf16x2(float a, float b) {
  v2bf p; p[0] = (__bf16)a; p[1] = (__bf16)b;
  return __builtin_bit_cast(unsigned int, p);
}
__device__ __forceinline__ v16bf as_v16bf(u32x8 x) {
  return __builtin_bit_cast(v16bf, x);
}
__device__ __forceinline__ v8f wmma_bf16(v16bf a, v16bf b, v8f c) {
  return __builtin_amdgcn_wmma_f32_16x16x32_bf16(false, a, false, b,
                                                 (short)0, c, false, false);
}

// Flat shared-memory address -> wave-relative LDS byte address (addr[31:0]).
__device__ __forceinline__ unsigned lds_addr32(const void* p) {
  return (unsigned)(size_t)p;
}
// 16B async global->LDS copy, GVS mode (SGPR64 base + per-lane byte offset).
__device__ __forceinline__ void async_copy_b128(unsigned lds_byte_addr,
                                                unsigned gl_byte_off,
                                                const void* base) {
  asm volatile("global_load_async_to_lds_b128 %0, %1, %2"
               :: "v"(lds_byte_addr), "v"(gl_byte_off), "s"(base)
               : "memory");
}
__device__ __forceinline__ void wait_async0() {
  asm volatile("s_wait_asynccnt 0x0" ::: "memory");
}

// ---------------------------------------------------------------------------
// Elementwise fp32 -> bf16 (packed stores). n must be a multiple of 4.
// ---------------------------------------------------------------------------
__global__ __launch_bounds__(256)
void cvt_f32_bf16_kernel(unsigned int* __restrict__ dst,
                         const float* __restrict__ src, long n4) {
  for (long i = (long)blockIdx.x * 256 + threadIdx.x; i < n4;
       i += (long)gridDim.x * 256) {
    float4 v = ((const float4*)src)[i];
    dst[i * 2]     = pack_bf16x2(v.x, v.y);
    dst[i * 2 + 1] = pack_bf16x2(v.z, v.w);
  }
}

// ---------------------------------------------------------------------------
// GEMM: C[M,N] (f32) = A[M,K] * W[N,K]^T, A/W already bf16 in global.
// 128x128 block, 8 waves (4 in M x 2 in N), wave tile 32x64 (2x4 WMMA tiles).
// Double-buffered LDS tiles filled by async global->LDS B128 copies.
// ---------------------------------------------------------------------------
__global__ __launch_bounds__(256)
void wmma_gemm_rt(float* __restrict__ C, const unsigned short* __restrict__ A,
                  const unsigned short* __restrict__ W, int M, int N, int K) {
  __shared__ unsigned int As[2][128][16];   // 2 x (128 rows x 32 bf16) = 16 KB
  __shared__ unsigned int Ws[2][128][16];   // 16 KB

  const int  tid  = (int)threadIdx.x;
  const int  lane = tid & 31;
  const int  wave = tid >> 5;
  const int  half = lane >> 4;
  const int  l16  = lane & 15;
  const int  wm   = wave >> 1;           // 0..3 (M strips of 32)
  const int  wn   = wave & 1;            // 0..1 (N strips of 64)
  const long bm   = (long)blockIdx.y * 128;
  const long bn   = (long)blockIdx.x * 128;

  // staging coordinates: 512 x 16B slots per tile, 2 per thread
  const int  srow0 = tid >> 2;           // 0..63
  const int  sq4   = tid & 3;            // 16B chunk within 64B row
  const long rA0 = bm + srow0, rA1 = bm + srow0 + 64;
  const long rW0 = bn + srow0, rW1 = bn + srow0 + 64;
  const bool vA0 = rA0 < M, vA1 = rA1 < M;
  const bool vW0 = rW0 < N, vW1 = rW1 < N;

  // issue the 4 async 16B copies this thread owns for K-tile k0 into buf.
  // Out-of-range rows are skipped: stale LDS only feeds unsaved accumulators.
  auto stage = [&](int buf, int k0) {
    if (vA0) async_copy_b128(lds_addr32(&As[buf][srow0][sq4 * 4]),
                             (unsigned)((rA0 * (long)K + k0 + sq4 * 8) * 2), A);
    if (vA1) async_copy_b128(lds_addr32(&As[buf][srow0 + 64][sq4 * 4]),
                             (unsigned)((rA1 * (long)K + k0 + sq4 * 8) * 2), A);
    if (vW0) async_copy_b128(lds_addr32(&Ws[buf][srow0][sq4 * 4]),
                             (unsigned)((rW0 * (long)K + k0 + sq4 * 8) * 2), W);
    if (vW1) async_copy_b128(lds_addr32(&Ws[buf][srow0 + 64][sq4 * 4]),
                             (unsigned)((rW1 * (long)K + k0 + sq4 * 8) * 2), W);
  };

  const v8f zero8 = {0.f,0.f,0.f,0.f,0.f,0.f,0.f,0.f};
  v8f acc[2][4];
#pragma unroll
  for (int mt = 0; mt < 2; ++mt)
#pragma unroll
    for (int nt = 0; nt < 4; ++nt) acc[mt][nt] = zero8;

  stage(0, 0);                           // prologue: tile 0 in flight
  int buf = 0;
  for (int k0 = 0; k0 < K; k0 += 32) {
    wait_async0();                       // our copies for `buf` have landed
    __syncthreads();                     // everyone's copies + prior reads done
    if (k0 + 32 < K) stage(buf ^ 1, k0 + 32);   // overlap next copy w/ compute

    // A fragments: lane m=l16; VGPR v<4 -> K=8h+2v, v>=4 -> K=16+8h+2(v-4)
    v16bf af[2];
#pragma unroll
    for (int mt = 0; mt < 2; ++mt) {
      const unsigned int* ap = &As[buf][wm * 32 + mt * 16 + l16][0];
      u32x8 f;
#pragma unroll
      for (int v = 0; v < 4; ++v) {
        f[v]     = ap[4 * half + v];
        f[v + 4] = ap[8 + 4 * half + v];
      }
      af[mt] = as_v16bf(f);
    }
    // B fragments: lane n=l16; VGPR v<4 -> K=8h+2v, v>=4 -> K=16+8h+2(v-4)
    v16bf bfm[4];
#pragma unroll
    for (int nt = 0; nt < 4; ++nt) {
      const unsigned int* wp = &Ws[buf][wn * 64 + nt * 16 + l16][0];
      u32x8 f;
#pragma unroll
      for (int v = 0; v < 8; ++v) f[v] = wp[(v < 4) ? (4 * half + v) : (8 + 4 * half + (v - 4))];
      bfm[nt] = as_v16bf(f);
    }
#pragma unroll
    for (int mt = 0; mt < 2; ++mt)
#pragma unroll
      for (int nt = 0; nt < 4; ++nt)
        acc[mt][nt] = wmma_bf16(af[mt], bfm[nt], acc[mt][nt]);
    buf ^= 1;
  }

  // C layout: slot r -> row r + 8*half, col = l16
#pragma unroll
  for (int mt = 0; mt < 2; ++mt)
#pragma unroll
    for (int nt = 0; nt < 4; ++nt)
#pragma unroll
      for (int r = 0; r < 8; ++r) {
        long row = bm + wm * 32 + mt * 16 + r + 8 * half;
        long col = bn + wn * 64 + nt * 16 + l16;
        if (row < M && col < N) C[row * N + col] = acc[mt][nt][r];
      }
}

// ---------------------------------------------------------------------------
// RMS norm, fp32 in -> bf16 out. One block per row.
// ---------------------------------------------------------------------------
__global__ __launch_bounds__(256)
void rmsnorm_kernel(unsigned short* __restrict__ dst,
                    const float* __restrict__ src,
                    const float* __restrict__ w, int cols, int src_stride,
                    int dst_stride) {
  const int row = blockIdx.x;
  const float* x = src + (size_t)row * src_stride;
  unsigned short* y = dst + (size_t)row * dst_stride;
  float ss = 0.f;
  for (int c = (int)threadIdx.x; c < cols; c += 256) { float v = x[c]; ss += v * v; }
#pragma unroll
  for (int m = 16; m >= 1; m >>= 1) ss += __shfl_xor(ss, m, 32);
  __shared__ float red[8];
  if ((threadIdx.x & 31) == 0) red[threadIdx.x >> 5] = ss;
  __syncthreads();
  float tot = 0.f;
#pragma unroll
  for (int i = 0; i < 8; ++i) tot += red[i];
  const float inv = rsqrtf(tot / (float)cols + 1e-6f);
  for (int c = (int)threadIdx.x; c < cols; c += 256)
    y[c] = bf16_bits(w[c] * x[c] * inv);
}

// ---------------------------------------------------------------------------
// RoPE + bf16 conversion of the full q row, plus roped k_pe (shared).
// Block = 544 threads (17 waves) per seq position.
// ---------------------------------------------------------------------------
__global__ __launch_bounds__(544)
void rope_cvt_kernel(unsigned short* __restrict__ qbf,
                     unsigned short* __restrict__ kpe,
                     const float* __restrict__ qf,
                     const float* __restrict__ kv_raw,
                     const int* __restrict__ pos_ids, int S) {
  const int srow = blockIdx.x;
  const int t    = (int)threadIdx.x;
  const int h    = t >> 5;
  const int j    = t & 31;
  const float p  = (float)pos_ids[srow];
  const float fr = __powf(10000.0f, -(float)j / 32.0f);
  float sn, cs;
  __sincosf(p * fr, &sn, &cs);
  if (h < 16) {
    const float*    src = qf  + (size_t)srow * 3072 + h * 192 + 128;
    unsigned short* d   = qbf + (size_t)srow * 3072 + h * 192 + 128;
    float x1 = src[j], x2 = src[j + 32];
    d[j]      = bf16_bits(x1 * cs - x2 * sn);
    d[j + 32] = bf16_bits(x2 * cs + x1 * sn);
  } else {
    const float* kp = kv_raw + (size_t)srow * 576 + 512;
    float x1 = kp[j], x2 = kp[j + 32];
    kpe[(size_t)srow * 64 + j]      = bf16_bits(x1 * cs - x2 * sn);
    kpe[(size_t)srow * 64 + j + 32] = bf16_bits(x2 * cs + x1 * sn);
  }
  for (int e = t; e < 2048; e += 544) {          // 16 heads x 128 nope dims
    int hh = e >> 7, col = e & 127;
    size_t idx = (size_t)srow * 3072 + hh * 192 + col;
    qbf[idx] = bf16_bits(qf[idx]);
  }
}

// ---------------------------------------------------------------------------
// Causal flash attention, one wave per (head, 16-query tile). All operands
// bf16 in global; fragments are direct uint loads of bf16 pairs.
// d_qk = 192 (128 nope + 64 shared roped pe), d_v = 128. bf16 output.
// ---------------------------------------------------------------------------
__global__ __launch_bounds__(32)
void mla_fa_kernel(unsigned short* __restrict__ O,
                   const unsigned short* __restrict__ Q,
                   const unsigned short* __restrict__ KV,
                   const unsigned short* __restrict__ KPE, int S) {
  const int head = blockIdx.y;
  const int q0   = blockIdx.x * 16;
  const int lane = (int)threadIdx.x;
  const int half = lane >> 4;
  const int l16  = lane & 15;
  const float scale = 0.072168783648703216f;    // 192^-0.5
  const v8f zero8 = {0.f,0.f,0.f,0.f,0.f,0.f,0.f,0.f};

  __shared__ unsigned int Pbuf[16][16];         // 16x32 bf16 P tile

  // --- Q fragments (A layout), 6 chunks of K=32 over 192 features ---
  v16bf qa[6];
  {
    const unsigned short* qrow = Q + (size_t)(q0 + l16) * 3072 + head * 192;
#pragma unroll
    for (int c = 0; c < 6; ++c) {
      u32x8 f;
#pragma unroll
      for (int v = 0; v < 4; ++v) {
        f[v]     = *(const unsigned int*)(qrow + c * 32 + half * 8 + 2 * v);
        f[v + 4] = *(const unsigned int*)(qrow + c * 32 + 16 + half * 8 + 2 * v);
      }
      qa[c] = as_v16bf(f);
    }
  }

  v8f oacc[8];
#pragma unroll
  for (int g = 0; g < 8; ++g) oacc[g] = zero8;
  float mrow[8], lrow[8];
#pragma unroll
  for (int r = 0; r < 8; ++r) { mrow[r] = -__builtin_inff(); lrow[r] = 0.f; }

  const int kmax = q0 + 15;                     // last causal key
  for (int kb = 0; kb <= kmax; kb += 32) {
    // ---- scores: two 16-key subtiles (12 WMMAs) ----
    v8f sc[2];
#pragma unroll
    for (int sub = 0; sub < 2; ++sub) {
      const int  key   = kb + sub * 16 + l16;
      const bool valid = key <= kmax;
      const unsigned short* krow_n = KV + (size_t)key * 4096 + head * 256;
      const unsigned short* krow_p = KPE + (size_t)key * 64;
      v8f s = zero8;
#pragma unroll
      for (int c = 0; c < 6; ++c) {
        const unsigned short* base =
            (c < 4) ? (krow_n + c * 32) : (krow_p + (c - 4) * 32);
        u32x8 f;
#pragma unroll
        for (int v = 0; v < 8; ++v)
          f[v] = valid ? *(const unsigned int*)(base + ((v < 4) ? (half * 8 + 2 * v) : (16 + half * 8 + 2 * (v - 4)))) : 0u;
        s = wmma_bf16(qa[c], as_v16bf(f), s);
      }
      sc[sub] = s;
    }

    // ---- online softmax (row r+8*half lives in slot r) ----
    __syncthreads();                            // previous P reads done
#pragma unroll
    for (int r = 0; r < 8; ++r) {
      const int qg = q0 + r + 8 * half;
      float s0 = sc[0][r] * scale;
      float s1 = sc[1][r] * scale;
      if (kb + l16 > qg)      s0 = -__builtin_inff();
      if (kb + 16 + l16 > qg) s1 = -__builtin_inff();
      float mx = fmaxf(s0, s1);
#pragma unroll
      for (int m = 8; m >= 1; m >>= 1) mx = fmaxf(mx, __shfl_xor(mx, m, 32));
      const float nm    = fmaxf(mrow[r], mx);
      const float alpha = __expf(mrow[r] - nm);
      mrow[r] = nm;
      const float p0 = __expf(s0 - nm);
      const float p1 = __expf(s1 - nm);
      float rs = p0 + p1;
#pragma unroll
      for (int m = 8; m >= 1; m >>= 1) rs += __shfl_xor(rs, m, 32);
      lrow[r] = lrow[r] * alpha + rs;
#pragma unroll
      for (int g = 0; g < 8; ++g) oacc[g][r] = oacc[g][r] * alpha;
      unsigned short* pr = (unsigned short*)&Pbuf[r + 8 * half][0];
      pr[l16]      = bf16_bits(p0);
      pr[16 + l16] = bf16_bits(p1);
    }
    __syncthreads();

    // ---- re-read P in A-fragment layout (uint pairs, v<4 -> K=8h+2v, v>=4 -> K=16+8h+2(v-4)) ----
    u32x8 pf;
#pragma unroll
    for (int v = 0; v < 4; ++v) {
      pf[v]     = Pbuf[l16][4 * half + v];
      pf[v + 4] = Pbuf[l16][8 + 4 * half + v];
    }
    const v16bf pA = as_v16bf(pf);

    // ---- P @ V (8 WMMAs; masked keys carry P==0) ----
#pragma unroll
    for (int g = 0; g < 8; ++g) {
      u32x8 f;
#pragma unroll
      for (int v = 0; v < 8; ++v) {
        const int key = kb + ((v < 4) ? (half * 8 + 2 * v) : (16 + half * 8 + 2 * (v - 4)));   // key <= S-1 (S % 32 == 0)
        const unsigned short* vp =
            KV + (size_t)key * 4096 + head * 256 + 128 + g * 16 + l16;
        f[v] = (unsigned int)vp[0] | ((unsigned int)vp[4096] << 16);
      }
      oacc[g] = wmma_bf16(pA, as_v16bf(f), oacc[g]);
    }
  }

  // ---- finalize: divide by row sums, bf16 store [S, head*128 + col] ----
#pragma unroll
  for (int r = 0; r < 8; ++r) {
    const int   row = q0 + r + 8 * half;
    const float inv = 1.f / lrow[r];
#pragma unroll
    for (int g = 0; g < 8; ++g)
      O[(size_t)row * 2048 + head * 128 + g * 16 + l16] =
          bf16_bits(oacc[g][r] * inv);
  }
}

// ---------------------------------------------------------------------------
static void launch_gemm(float* C, const unsigned short* A,
                        const unsigned short* W, int M, int N, int K,
                        hipStream_t s) {
  dim3 grid((N + 127) / 128, (M + 127) / 128);
  wmma_gemm_rt<<<grid, dim3(256), 0, s>>>(C, A, W, M, N, K);
}
static void launch_cvt(unsigned short* dst, const float* src, long n,
                       hipStream_t s) {
  long n4 = n / 4;
  int grid = (int)((n4 + 255) / 256);
  if (grid > 8192) grid = 8192;
  cvt_f32_bf16_kernel<<<grid, 256, 0, s>>>((unsigned int*)dst, src, n4);
}

extern "C" void kernel_launch(void* const* d_in, const int* in_sizes, int n_in,
                              void* d_out, int out_size, void* d_ws, size_t ws_size,
                              hipStream_t stream) {
  (void)n_in; (void)out_size; (void)ws_size;
  const float* hidden    = (const float*)d_in[0];   // [1,S,4096]
  const int*   pos_ids   = (const int*)d_in[1];     // [1,S]
  const float* wq_a      = (const float*)d_in[2];   // [1536,4096]
  const float* q_a_norm  = (const float*)d_in[3];   // [1536]
  const float* wq_b      = (const float*)d_in[4];   // [3072,1536]
  const float* wkv_a     = (const float*)d_in[5];   // [576,4096]
  const float* kv_a_norm = (const float*)d_in[6];   // [512]
  const float* wkv_b     = (const float*)d_in[7];   // [4096,512]
  const float* wo        = (const float*)d_in[8];   // [4096,2048]
  float* out = (float*)d_out;                       // [S,4096]

  const int S = in_sizes[0] / 4096;

  // --- workspace: f32 region then bf16 region (~190 MB total for S=2048) ---
  float* wsf    = (float*)d_ws;
  float* q_lat  = wsf; wsf += (size_t)S * 1536;     // h@wq_a^T (pre-norm)
  float* kv_raw = wsf; wsf += (size_t)S * 576;      // h@wkv_a^T
  float* q_f    = wsf; wsf += (size_t)S * 3072;     // q_lat@wq_b^T (pre-rope)
  float* kv_exp = wsf; wsf += (size_t)S * 4096;     // ckv@wkv_b^T

  unsigned short* wsh = (unsigned short*)wsf;
  unsigned short* hid_bf   = wsh; wsh += (size_t)S * 4096;
  unsigned short* wqa_bf   = wsh; wsh += (size_t)1536 * 4096;
  unsigned short* wqb_bf   = wsh; wsh += (size_t)3072 * 1536;
  unsigned short* wkva_bf  = wsh; wsh += (size_t)576 * 4096;
  unsigned short* wkvb_bf  = wsh; wsh += (size_t)4096 * 512;
  unsigned short* wo_bf    = wsh; wsh += (size_t)4096 * 2048;
  unsigned short* qlat_bf  = wsh; wsh += (size_t)S * 1536;
  unsigned short* ckv_bf   = wsh; wsh += (size_t)S * 512;
  unsigned short* q_bf     = wsh; wsh += (size_t)S * 3072;
  unsigned short* kvexp_bf = wsh; wsh += (size_t)S * 4096;
  unsigned short* kpe_bf   = wsh; wsh += (size_t)S * 64;
  unsigned short* attn_bf  = wsh; wsh += (size_t)S * 2048;

  // one-time bf16 conversions (inputs + weights)
  launch_cvt(hid_bf,  hidden, (long)S * 4096,     stream);
  launch_cvt(wqa_bf,  wq_a,   (long)1536 * 4096,  stream);
  launch_cvt(wqb_bf,  wq_b,   (long)3072 * 1536,  stream);
  launch_cvt(wkva_bf, wkv_a,  (long)576 * 4096,   stream);
  launch_cvt(wkvb_bf, wkv_b,  (long)4096 * 512,   stream);
  launch_cvt(wo_bf,   wo,     (long)4096 * 2048,  stream);

  // pipeline
  launch_gemm(q_lat,  hid_bf, wqa_bf,  S, 1536, 4096, stream);
  launch_gemm(kv_raw, hid_bf, wkva_bf, S,  576, 4096, stream);
  rmsnorm_kernel<<<S, 256, 0, stream>>>(qlat_bf, q_lat,  q_a_norm,  1536, 1536, 1536);
  rmsnorm_kernel<<<S, 256, 0, stream>>>(ckv_bf,  kv_raw, kv_a_norm,  512,  576,  512);
  launch_gemm(q_f,    qlat_bf, wqb_bf,  S, 3072, 1536, stream);
  launch_gemm(kv_exp, ckv_bf,  wkvb_bf, S, 4096,  512, stream);
  launch_cvt(kvexp_bf, kv_exp, (long)S * 4096, stream);
  rope_cvt_kernel<<<S, 544, 0, stream>>>(q_bf, kpe_bf, q_f, kv_raw, pos_ids, S);
  mla_fa_kernel<<<dim3(S / 16, 16), 32, 0, stream>>>(attn_bf, q_bf, kvexp_bf,
                                                     kpe_bf, S);
  launch_gemm(out, attn_bf, wo_bf, S, 4096, 2048, stream);
}
